// HybridAttentionMILold_34703335752340
// MI455X (gfx1250) — matrix operand order repaired and checked by reading, not compiled, not run on hardware
//
#include <hip/hip_runtime.h>
#include <stdint.h>

// Problem constants (match reference)
#define N_BAG   500000
#define DIM     128
#define HID     128
#define GDIM    64
#define MDIM    256
#define NTILES  (N_BAG / 16)     // 31250 row-tiles of 16 instances
#define NPAIRS  (NTILES / 2)     // 15625 pair-tiles (32 rows each)
#define K1_BLOCKS 512            // 512 blocks * 8 waves = 4096 waves

typedef __bf16 bf16;
typedef __attribute__((ext_vector_type(16))) bf16  v16bf;
typedef __attribute__((ext_vector_type(8)))  float v8f;

// ---------------------------------------------------------------------------
// B fragment (32x16 bf16, KxN): lane n<16 -> column h, elements 0..15 = K 0..15
// of this 32-chunk; lanes 16..31 -> K 16..31. 32 contiguous bytes per lane.
// ---------------------------------------------------------------------------
__device__ __forceinline__ v16bf load_bfrag(const bf16* __restrict__ wb,
                                            int h, int kk, int lh) {
  const bf16* p = wb + h * DIM + kk * 32 + lh * 16;
  union { uint4 u[2]; v16bf v; } fr;
  fr.u[0] = *(const uint4*)(p);
  fr.u[1] = *(const uint4*)(p + 8);
  return fr.v;
}

// ---------------------------------------------------------------------------
// A fragment (16x32 bf16, MxK): lane m<16 -> row, elems 0..15 = K+0..15;
// lanes 16..31 -> K+16..31.
// ---------------------------------------------------------------------------
__device__ __forceinline__ v16bf make_afrag(const float* __restrict__ rowp, int c0) {
  const float4 f0 = *(const float4*)(rowp + c0);
  const float4 f1 = *(const float4*)(rowp + c0 + 4);
  const float4 f2 = *(const float4*)(rowp + c0 + 8);
  const float4 f3 = *(const float4*)(rowp + c0 + 12);
  v16bf a;
  a[0] = (bf16)f0.x; a[1] = (bf16)f0.y; a[2]  = (bf16)f0.z; a[3]  = (bf16)f0.w;
  a[4] = (bf16)f1.x; a[5] = (bf16)f1.y; a[6]  = (bf16)f1.z; a[7]  = (bf16)f1.w;
  a[8] = (bf16)f2.x; a[9] = (bf16)f2.y; a[10] = (bf16)f2.z; a[11] = (bf16)f2.w;
  a[12]= (bf16)f3.x; a[13]= (bf16)f3.y; a[14] = (bf16)f3.z; a[15] = (bf16)f3.w;
  return a;
}

// ---------------------------------------------------------------------------
// K0: convert Wv/Wu fp32 -> bf16 (row-major [H][D]); resident in L2 afterwards
// ---------------------------------------------------------------------------
__global__ __launch_bounds__(256) void k0_cvt(const float* __restrict__ Wv,
                                              const float* __restrict__ Wu,
                                              bf16* __restrict__ wvb,
                                              bf16* __restrict__ wub) {
  int i = blockIdx.x * blockDim.x + threadIdx.x;
  if (i < HID * DIM) {
    wvb[i] = (bf16)Wv[i];
    wub[i] = (bf16)Wu[i];
  }
}

// Online-softmax update for one 16-row sub-tile; RACC[s] holds A_raw for rows
// (TBASE + s + 8*lh), replicated across each 16-lane half. Kept as a macro so
// the register arrays stay SROA-able (no scratch).
#define ONLINE_UPDATE(RACC, TBASE)                                             \
  do {                                                                         \
    if (sub == 0) {                                                            \
      _Pragma("unroll")                                                        \
      for (int s = 0; s < 8; ++s) araw[(TBASE) + 8 * lh + s] = RACC[s];        \
    }                                                                          \
    float mloc = RACC[0];                                                      \
    _Pragma("unroll")                                                          \
    for (int s = 1; s < 8; ++s) mloc = fmaxf(mloc, RACC[s]);                   \
    mloc = fmaxf(mloc, __shfl_xor(mloc, 16));                                  \
    const float newm  = fmaxf(m_w, mloc);                                      \
    const float scale = __expf(m_w - newm);                                    \
    float es[8]; float ssum = 0.f;                                             \
    _Pragma("unroll")                                                          \
    for (int s = 0; s < 8; ++s) { es[s] = __expf(RACC[s] - newm); ssum += es[s]; } \
    ssum += __shfl_xor(ssum, 16);                                              \
    s_w = s_w * scale + ssum;                                                  \
    e0 *= scale; e1 *= scale; e2 *= scale; e3 *= scale;                        \
    _Pragma("unroll")                                                          \
    for (int s = 0; s < 8; ++s) {                                              \
      const float wlo = __shfl(es[s], 0);                                      \
      const float whi = __shfl(es[s], 16);                                     \
      const float4 blo = *(const float4*)(bag + (size_t)((TBASE) + s)     * DIM + 4 * lane); \
      const float4 bhi = *(const float4*)(bag + (size_t)((TBASE) + s + 8) * DIM + 4 * lane); \
      e0 += wlo * blo.x + whi * bhi.x;                                         \
      e1 += wlo * blo.y + whi * bhi.y;                                         \
      e2 += wlo * blo.z + whi * bhi.z;                                         \
      e3 += wlo * blo.w + whi * bhi.w;                                         \
    }                                                                          \
    m_w = newm;                                                                \
  } while (0)

// ---------------------------------------------------------------------------
// K1: fused attention scores (WMMA) + online-softmax weighted embedding.
// Each wave owns pair-tiles of 32 rows; one B fragment feeds 2 WMMAs.
// ---------------------------------------------------------------------------
__global__ __launch_bounds__(256) void k1_attn(
    const float* __restrict__ bag,
    const float* __restrict__ bv, const float* __restrict__ bu,
    const float* __restrict__ ww, const float* __restrict__ bwp,
    const bf16* __restrict__ wvb, const bf16* __restrict__ wub,
    float* __restrict__ araw,
    float* __restrict__ blkM, float* __restrict__ blkS,
    float* __restrict__ blkE) {
  __shared__ float sM[8], sS[8], sE[8][DIM];
  __shared__ float sBM;

  const int lane = threadIdx.x & 31;
  const int wid  = threadIdx.x >> 5;
  const int sub  = lane & 15;     // row (A) / column (B) within tile
  const int lh   = lane >> 4;     // lane half: selects K sub-block
  const int gwave  = blockIdx.x * 8 + wid;
  const int nwaves = K1_BLOCKS * 8;
  const float bw0 = bwp[0];

  // Per-lane h-dependent constants for each of the 8 h-tiles (hoisted, L1 hits)
  float bvh[8], buh[8], wwh[8];
#pragma unroll
  for (int ht = 0; ht < 8; ++ht) {
    const int h = ht * 16 + sub;
    bvh[ht] = bv[h]; buh[ht] = bu[h]; wwh[ht] = ww[h];
  }

  // online-softmax state (uniform across lanes except e*)
  float m_w = -__builtin_inff();
  float s_w = 0.f;
  float e0 = 0.f, e1 = 0.f, e2 = 0.f, e3 = 0.f;   // cols 4*lane .. 4*lane+3

  int zoff = 0;  // opaque zero: re-defined each iteration to defeat LICM so the
                 // B-fragment loads stay inside the loop (L1 hits, no spills)

  for (int pt = gwave; pt < NPAIRS; pt += nwaves) {
    asm volatile("" : "+s"(zoff));
    const bf16* wv_p = wvb + zoff;
    const bf16* wu_p = wub + zoff;

    const int tb0 = pt * 32;        // rows of sub-tile 0
    const int tb1 = tb0 + 16;       // rows of sub-tile 1
    const float* arow0 = bag + (size_t)(tb0 + sub) * DIM;
    const float* arow1 = bag + (size_t)(tb1 + sub) * DIM;

    // A fragments for both sub-tiles, K = 0..127 (4 chunks of 32)
    v16bf aF0[4], aF1[4];
#pragma unroll
    for (int kk = 0; kk < 4; ++kk) {
      aF0[kk] = make_afrag(arow0, kk * 32 + lh * 16);
      aF1[kk] = make_afrag(arow1, kk * 32 + lh * 16);
    }

    float racc0[8], racc1[8];
#pragma unroll
    for (int s = 0; s < 8; ++s) { racc0[s] = 0.f; racc1[s] = 0.f; }

#pragma unroll
    for (int ht = 0; ht < 8; ++ht) {
      v8f aV0, aU0, aV1, aU1;
#pragma unroll
      for (int s = 0; s < 8; ++s) {
        aV0[s] = bvh[ht]; aU0[s] = buh[ht];
        aV1[s] = bvh[ht]; aU1[s] = buh[ht];
      }
#pragma unroll
      for (int kk = 0; kk < 4; ++kk) {
        const v16bf bV = load_bfrag(wv_p, ht * 16 + sub, kk, lh);
        const v16bf bU = load_bfrag(wu_p, ht * 16 + sub, kk, lh);
        aV0 = __builtin_amdgcn_wmma_f32_16x16x32_bf16(false, aF0[kk], false, bV,
                                                      (short)0, aV0, false, false);
        aU0 = __builtin_amdgcn_wmma_f32_16x16x32_bf16(false, aF0[kk], false, bU,
                                                      (short)0, aU0, false, false);
        aV1 = __builtin_amdgcn_wmma_f32_16x16x32_bf16(false, aF1[kk], false, bV,
                                                      (short)0, aV1, false, false);
        aU1 = __builtin_amdgcn_wmma_f32_16x16x32_bf16(false, aF1[kk], false, bU,
                                                      (short)0, aU1, false, false);
      }
      // gated attention: tanh(V) * sigmoid(U) * Ww[h], summed over h
#pragma unroll
      for (int s = 0; s < 8; ++s) {
        float t  = __expf(2.f * aV0[s]);
        float vt = (t - 1.f) / (t + 1.f);
        float u  = 1.f / (1.f + __expf(-aU0[s]));
        racc0[s] += vt * u * wwh[ht];
        t  = __expf(2.f * aV1[s]);
        vt = (t - 1.f) / (t + 1.f);
        u  = 1.f / (1.f + __expf(-aU1[s]));
        racc1[s] += vt * u * wwh[ht];
      }
    }

    // reduce over the 16 h-columns inside each lane-half (wave32)
#pragma unroll
    for (int s = 0; s < 8; ++s) {
      racc0[s] += __shfl_xor(racc0[s], 1);
      racc0[s] += __shfl_xor(racc0[s], 2);
      racc0[s] += __shfl_xor(racc0[s], 4);
      racc0[s] += __shfl_xor(racc0[s], 8);
      racc0[s] += bw0;
      racc1[s] += __shfl_xor(racc1[s], 1);
      racc1[s] += __shfl_xor(racc1[s], 2);
      racc1[s] += __shfl_xor(racc1[s], 4);
      racc1[s] += __shfl_xor(racc1[s], 8);
      racc1[s] += bw0;
    }

    ONLINE_UPDATE(racc0, tb0);
    ONLINE_UPDATE(racc1, tb1);
  }

  // ---- block combine: 8 waves -> one (m, s, e[128]) partial ----
  sE[wid][4 * lane + 0] = e0;
  sE[wid][4 * lane + 1] = e1;
  sE[wid][4 * lane + 2] = e2;
  sE[wid][4 * lane + 3] = e3;
  if (lane == 0) { sM[wid] = m_w; sS[wid] = s_w; }
  __syncthreads();
  if (threadIdx.x == 0) {
    float bm = sM[0];
#pragma unroll
    for (int w = 1; w < 8; ++w) bm = fmaxf(bm, sM[w]);
    sBM = bm;
  }
  __syncthreads();
  const float bm = sBM;
  if (threadIdx.x < DIM) {
    float acc = 0.f;
#pragma unroll
    for (int w = 0; w < 8; ++w) acc += __expf(sM[w] - bm) * sE[w][threadIdx.x];
    blkE[blockIdx.x * DIM + threadIdx.x] = acc;
  }
  if (threadIdx.x == 0) {
    float ss = 0.f;
#pragma unroll
    for (int w = 0; w < 8; ++w) ss += __expf(sM[w] - bm) * sS[w];
    blkS[blockIdx.x] = ss;
    blkM[blockIdx.x] = bm;
  }
}

// ---------------------------------------------------------------------------
// K2: combine block partials -> global (m, 1/S), embedding; regressor head.
// ---------------------------------------------------------------------------
__global__ __launch_bounds__(256) void k2_head(
    const float* __restrict__ blkM, const float* __restrict__ blkS,
    const float* __restrict__ blkE, int nb,
    const float* __restrict__ gfeat,
    const float* __restrict__ W1, const float* __restrict__ b1,
    const float* __restrict__ W2, const float* __restrict__ b2,
    float* __restrict__ head, float* __restrict__ out_score) {
  __shared__ float fused[DIM + GDIM];
  __shared__ float red[MDIM];
  __shared__ float sm2[2];
  const int t = threadIdx.x;

  if (t == 0) {
    float m = -__builtin_inff();
    for (int b = 0; b < nb; ++b) m = fmaxf(m, blkM[b]);
    float S = 0.f;
    for (int b = 0; b < nb; ++b) S += __expf(blkM[b] - m) * blkS[b];
    const float invS = 1.0f / S;
    sm2[0] = m; sm2[1] = invS;
    head[0] = m; head[1] = invS;
  }
  __syncthreads();
  const float m = sm2[0], invS = sm2[1];

  if (t < DIM) {
    float e = 0.f;
    for (int b = 0; b < nb; ++b) e += __expf(blkM[b] - m) * blkE[b * DIM + t];
    fused[t] = e * invS;                 // bag_embedding
  } else if (t < DIM + GDIM) {
    fused[t] = gfeat[t - DIM];           // concat global features
  }
  __syncthreads();

  float h = b1[t];
  for (int k = 0; k < DIM + GDIM; ++k) h += fused[k] * W1[t * (DIM + GDIM) + k];
  h = (h > 0.f) ? h : 0.01f * h;         // LeakyReLU(0.01)
  red[t] = h * W2[t];
  __syncthreads();
  for (int s = MDIM / 2; s > 0; s >>= 1) {
    if (t < s) red[t] += red[t + s];
    __syncthreads();
  }
  if (t == 0) out_score[0] = red[0] + b2[0];
}

// ---------------------------------------------------------------------------
// K3: normalize alpha in place: alpha[n] = exp(Araw[n] - m) / S
// ---------------------------------------------------------------------------
__global__ __launch_bounds__(256) void k3_alpha(float* __restrict__ alpha,
                                                const float* __restrict__ head,
                                                int n) {
  const int i = blockIdx.x * blockDim.x + threadIdx.x;
  if (i < n) {
    const float m = head[0], invS = head[1];
    alpha[i] = __expf(alpha[i] - m) * invS;
  }
}

// ---------------------------------------------------------------------------
extern "C" void kernel_launch(void* const* d_in, const int* in_sizes, int n_in,
                              void* d_out, int out_size, void* d_ws, size_t ws_size,
                              hipStream_t stream) {
  const float* bag   = (const float*)d_in[0];   // [N, D]
  const float* gfeat = (const float*)d_in[1];   // [G]
  const float* Wv    = (const float*)d_in[2];   // [H, D]
  const float* bv    = (const float*)d_in[3];   // [H]
  const float* Wu    = (const float*)d_in[4];   // [H, D]
  const float* bu    = (const float*)d_in[5];   // [H]
  const float* Ww    = (const float*)d_in[6];   // [1, H]
  const float* bw    = (const float*)d_in[7];   // [1]
  const float* W1    = (const float*)d_in[8];   // [M, D+G]
  const float* b1    = (const float*)d_in[9];   // [M]
  const float* W2    = (const float*)d_in[10];  // [1, M]
  const float* b2    = (const float*)d_in[11];  // [1]

  float* out = (float*)d_out;         // [0] = score, [1..N] = alpha
  float* araw = out + 1;              // raw scores staged in alpha's slot

  // workspace layout (~324 KB total)
  char* ws = (char*)d_ws;
  bf16*  wvb  = (bf16*)(ws);                               // 32 KB
  bf16*  wub  = (bf16*)(ws + 32768);                       // 32 KB
  float* blkM = (float*)(ws + 65536);                      // 2 KB
  float* blkS = (float*)(ws + 65536 + 2048);               // 2 KB
  float* blkE = (float*)(ws + 65536 + 4096);               // 256 KB
  float* head = (float*)(ws + 65536 + 4096 + K1_BLOCKS * DIM * 4);

  k0_cvt<<<(HID * DIM + 255) / 256, 256, 0, stream>>>(Wv, Wu, wvb, wub);

  k1_attn<<<K1_BLOCKS, 256, 0, stream>>>(bag, bv, bu, Ww, bw, wvb, wub,
                                         araw, blkM, blkS, blkE);

  k2_head<<<1, MDIM, 0, stream>>>(blkM, blkS, blkE, K1_BLOCKS, gfeat,
                                  W1, b1, W2, b2, head, out);

  k3_alpha<<<(N_BAG + 255) / 256, 256, 0, stream>>>(araw, head, N_BAG);
}
